// Attention_12137577578573
// MI455X (gfx1250) — matrix operand order repaired and checked by reading, not compiled, not run on hardware
//
#include <hip/hip_runtime.h>

typedef __attribute__((ext_vector_type(16))) _Float16 v16h;
typedef __attribute__((ext_vector_type(8)))  _Float16 v8h;
typedef __attribute__((ext_vector_type(2)))  _Float16 v2h;
typedef __attribute__((ext_vector_type(8)))  float    v8f;
typedef __attribute__((ext_vector_type(4)))  float    v4f;

#define B_   4
#define F_   2048
#define T_   2048
#define C_   1024     // CF == CT
#define H_   16
#define DH_  64
#define HD_  1024     // H*DH
#define ALPHA_ 0.125f // 1/sqrt(64)

union hv16 { v16h v; v8h h[2]; };

// ---------------------------------------------------------------------------
// Kernel 1: fused QKV projection.  X[8192,1024] @ W[1024,1024] + b, f32 in,
// f16 out, scattered into WMMA-friendly layouts:
//   proj 0 (Q): Qh[b][h][f][d]          (c = h*64 + d)
//   proj 1 (K): Kh[b][h][t][d]          (c = d*16 + h   <- reference reshape)
//   proj 2 (V): Vt[b][h][d][t]          (transposed for PV B-fragments)
// Block: 256 thr = 8 waves; tile 64(M) x 128(N); K staged in 32-chunks in LDS
// with a register-staged software pipeline (global prefetch of chunk k+32
// overlaps the WMMAs consuming chunk k).
// ---------------------------------------------------------------------------
__global__ __launch_bounds__(256) void qkv_proj_kernel(
    const float* __restrict__ from_t, const float* __restrict__ to_t,
    const float* __restrict__ Wq, const float* __restrict__ bq,
    const float* __restrict__ Wk, const float* __restrict__ bk,
    const float* __restrict__ Wv, const float* __restrict__ bv,
    _Float16* __restrict__ Qh, _Float16* __restrict__ Kh,
    _Float16* __restrict__ Vt)
{
  __shared__ _Float16 Xs[64 * 32];    // [m][k]   row-major
  __shared__ _Float16 WsT[128 * 32];  // [n][k]   transposed

  const int proj  = blockIdx.z;
  const int mbase = blockIdx.y * 64;
  const int nbase = blockIdx.x * 128;
  const int tid   = threadIdx.x;
  const int lane  = tid & 31;
  const int w     = tid >> 5;
  const int wm    = w & 3;     // M sub-tile (16 rows)
  const int wn    = w >> 2;    // N sub-tile (64 cols)

  const float* X    = (proj == 0) ? from_t : to_t;
  const float* W    = (proj == 0) ? Wq : (proj == 1) ? Wk : Wv;
  const float* bias = (proj == 0) ? bq : (proj == 1) ? bk : bv;

  v8f acc[4] = {v8f{}, v8f{}, v8f{}, v8f{}};

  const int xrow = tid >> 2;        // 0..63
  const int xcol = (tid & 3) * 8;   // 0,8,16,24
  const int wn_  = tid & 127;       // W column within tile
  const int wkb  = (tid >> 7) * 16; // k sub-range base: 0 or 16

  const int am  = lane & 15;
  const int ac  = (lane >> 4) * 16;
  const int bn  = lane & 15;
  const int bk2 = (lane >> 4) * 16;

  // ---- software pipeline registers ----
  v4f   xa, xb;
  float wreg[16];

  // preload chunk 0
  {
    const float* xp = X + (size_t)(mbase + xrow) * C_ + xcol;
    xa = *(const v4f*)xp;
    xb = *(const v4f*)(xp + 4);
#pragma unroll
    for (int i = 0; i < 16; ++i)
      wreg[i] = W[(size_t)(wkb + i) * HD_ + nbase + wn_];
  }

  for (int kc = 0; kc < C_; kc += 32) {
    // ---- commit staged registers to LDS (f32 -> f16) ----
    {
      v8h xh;
      xh[0] = (_Float16)xa[0]; xh[1] = (_Float16)xa[1];
      xh[2] = (_Float16)xa[2]; xh[3] = (_Float16)xa[3];
      xh[4] = (_Float16)xb[0]; xh[5] = (_Float16)xb[1];
      xh[6] = (_Float16)xb[2]; xh[7] = (_Float16)xb[3];
      *(v8h*)&Xs[xrow * 32 + xcol] = xh;
#pragma unroll
      for (int i = 0; i < 8; ++i) {
        v2h p;
        p[0] = (_Float16)wreg[2 * i];
        p[1] = (_Float16)wreg[2 * i + 1];
        *(v2h*)&WsT[wn_ * 32 + wkb + 2 * i] = p;
      }
    }
    __syncthreads();

    // ---- issue global prefetch of next chunk (overlaps with WMMAs) ----
    if (kc + 32 < C_) {
      const float* xp = X + (size_t)(mbase + xrow) * C_ + (kc + 32) + xcol;
      xa = *(const v4f*)xp;
      xb = *(const v4f*)(xp + 4);
#pragma unroll
      for (int i = 0; i < 16; ++i)
        wreg[i] = W[(size_t)(kc + 32 + wkb + i) * HD_ + nbase + wn_];
    }

    // ---- compute: 1 A fragment, 4 B fragments, 4 back-to-back WMMAs ----
    hv16 A;
    A.h[0] = *(const v8h*)&Xs[(wm * 16 + am) * 32 + ac];
    A.h[1] = *(const v8h*)&Xs[(wm * 16 + am) * 32 + 8 + ac];
    const v16h Bf0 = *(const v16h*)&WsT[(wn * 64 +  0 + bn) * 32 + bk2];
    const v16h Bf1 = *(const v16h*)&WsT[(wn * 64 + 16 + bn) * 32 + bk2];
    const v16h Bf2 = *(const v16h*)&WsT[(wn * 64 + 32 + bn) * 32 + bk2];
    const v16h Bf3 = *(const v16h*)&WsT[(wn * 64 + 48 + bn) * 32 + bk2];
    acc[0] = __builtin_amdgcn_wmma_f32_16x16x32_f16(false, A.v, false, Bf0,
                                                    (short)0, acc[0], false, false);
    acc[1] = __builtin_amdgcn_wmma_f32_16x16x32_f16(false, A.v, false, Bf1,
                                                    (short)0, acc[1], false, false);
    acc[2] = __builtin_amdgcn_wmma_f32_16x16x32_f16(false, A.v, false, Bf2,
                                                    (short)0, acc[2], false, false);
    acc[3] = __builtin_amdgcn_wmma_f32_16x16x32_f16(false, A.v, false, Bf3,
                                                    (short)0, acc[3], false, false);
    __syncthreads();
  }

  // ---- epilogue: bias + scatter into attention layouts ----
  const int M0 = (lane >> 4) * 8;
  const int nn = lane & 15;
#pragma unroll
  for (int j = 0; j < 4; ++j) {
    const int col = nbase + wn * 64 + j * 16 + nn;
    const float bb = bias[col];
#pragma unroll
    for (int r = 0; r < 8; ++r) {
      const int row = mbase + wm * 16 + M0 + r;
      const int b = row >> 11;      // / 2048
      const int f = row & 2047;
      const _Float16 hv = (_Float16)(acc[j][r] + bb);
      if (proj == 0) {
        const int hh = col >> 6, d = col & 63;
        Qh[(((size_t)b * H_ + hh) * F_ + f) * DH_ + d] = hv;
      } else if (proj == 1) {
        const int hh = col & 15, d = col >> 4;   // reference: c = d*H + h
        Kh[(((size_t)b * H_ + hh) * T_ + f) * DH_ + d] = hv;
      } else {
        const int hh = col >> 6, d = col & 63;
        Vt[(((size_t)b * H_ + hh) * DH_ + d) * T_ + f] = hv;
      }
    }
  }
}

// ---------------------------------------------------------------------------
// Kernel 2: flash attention.  grid = (F/128, B*H); 8 waves/block, each wave
// owns 16 F-rows of one (b,h); loops T in 32-wide chunks with online softmax.
// Load issue order per chunk: K (needed first) -> mask -> V (needed last),
// so every load's latency hides behind WMMA / softmax VALU work.
// ---------------------------------------------------------------------------
__global__ __launch_bounds__(256) void flash_attn_kernel(
    const _Float16* __restrict__ Qh, const _Float16* __restrict__ Kh,
    const _Float16* __restrict__ Vt, const int* __restrict__ mask,
    float* __restrict__ out)
{
  __shared__ _Float16 Pbuf[8][16 * 32];   // per-wave P bounce tile (1KB each)

  const int tid  = threadIdx.x;
  const int lane = tid & 31;
  const int w    = tid >> 5;
  const int bh   = blockIdx.y;
  const int b    = bh >> 4;
  const int h    = bh & 15;
  const int fbase = blockIdx.x * 128 + w * 16;

  const _Float16* Qbh = Qh + ((size_t)b * H_ + h) * F_ * DH_;
  const _Float16* Kbh = Kh + ((size_t)b * H_ + h) * T_ * DH_;
  const _Float16* Vbh = Vt + ((size_t)b * H_ + h) * DH_ * T_;
  const int* mbase_p  = mask + (size_t)b * F_ * T_;

  const int nn = lane & 15;
  const int hi = lane >> 4;          // 0/1 half-wave
  _Float16* P = Pbuf[w];

  // Q A-fragments, held for the whole T loop: d 0..31 and 32..63
  hv16 A0, A1;
  {
    const int m = lane & 15;
    const int c = hi * 16;
    const _Float16* qrow = Qbh + (size_t)(fbase + m) * DH_;
    A0.h[0] = *(const v8h*)(qrow + c);
    A0.h[1] = *(const v8h*)(qrow + 8 + c);
    A1.h[0] = *(const v8h*)(qrow + 32 + c);
    A1.h[1] = *(const v8h*)(qrow + 40 + c);
  }

  v8f O0 = {}, O1 = {}, O2 = {}, O3 = {};
  float mrow[8], lrow[8];
#pragma unroll
  for (int r = 0; r < 8; ++r) { mrow[r] = -1e30f; lrow[r] = 0.f; }

  for (int t0 = 0; t0 < T_; t0 += 32) {
    // ---- issue K B-fragment loads (consumed first, by the S WMMAs) ----
    const _Float16* k0 = Kbh + (size_t)(t0 + nn) * DH_;
    const _Float16* k1 = Kbh + (size_t)(t0 + 16 + nn) * DH_;
    const v16h B00 = *(const v16h*)(k0 + hi * 16);
    const v16h B01 = *(const v16h*)(k0 + 32 + hi * 16);
    const v16h B10 = *(const v16h*)(k1 + hi * 16);
    const v16h B11 = *(const v16h*)(k1 + 32 + hi * 16);

    // ---- issue mask loads (consumed after S WMMAs) ----
    float ad0[8], ad1[8];
#pragma unroll
    for (int r = 0; r < 8; ++r) {
      const int row = fbase + hi * 8 + r;
      const int* mr = mbase_p + (size_t)row * T_ + t0;
      ad0[r] = (1.f - (float)mr[nn])      * -100000.f;
      ad1[r] = (1.f - (float)mr[16 + nn]) * -100000.f;
    }

    // ---- issue V B-fragment loads (consumed last, by the PV WMMAs) ----
    const size_t tcol = (size_t)t0 + hi * 16;
    const v16h VB0 = *(const v16h*)(Vbh + (size_t)(0 * 16 + nn) * T_ + tcol);
    const v16h VB1 = *(const v16h*)(Vbh + (size_t)(1 * 16 + nn) * T_ + tcol);
    const v16h VB2 = *(const v16h*)(Vbh + (size_t)(2 * 16 + nn) * T_ + tcol);
    const v16h VB3 = *(const v16h*)(Vbh + (size_t)(3 * 16 + nn) * T_ + tcol);

    if (t0 + 32 < T_) {  // prefetch next K / V chunks
      __builtin_prefetch(Kbh + (size_t)(t0 + 32 + nn) * DH_, 0, 1);
      __builtin_prefetch(Vbh + (size_t)nn * T_ + t0 + 32, 0, 1);
    }

    // ---- S = Q K^T (two 16-col tiles, K-dim = DH split 2x32) ----
    v8f S0 = {}, S1 = {};
    S0 = __builtin_amdgcn_wmma_f32_16x16x32_f16(false, A0.v, false, B00,
                                                (short)0, S0, false, false);
    S0 = __builtin_amdgcn_wmma_f32_16x16x32_f16(false, A1.v, false, B01,
                                                (short)0, S0, false, false);
    S1 = __builtin_amdgcn_wmma_f32_16x16x32_f16(false, A0.v, false, B10,
                                                (short)0, S1, false, false);
    S1 = __builtin_amdgcn_wmma_f32_16x16x32_f16(false, A1.v, false, B11,
                                                (short)0, S1, false, false);

    // ---- scale + additive mask ----
#pragma unroll
    for (int r = 0; r < 8; ++r) {
      S0[r] = S0[r] * ALPHA_ + ad0[r];
      S1[r] = S1[r] * ALPHA_ + ad1[r];
    }

    // ---- online softmax (row lives in 16 lanes of one half-wave) ----
    float cf[8];
#pragma unroll
    for (int r = 0; r < 8; ++r) {
      float mx = fmaxf(S0[r], S1[r]);
      mx = fmaxf(mx, __shfl_xor(mx, 1, 32));
      mx = fmaxf(mx, __shfl_xor(mx, 2, 32));
      mx = fmaxf(mx, __shfl_xor(mx, 4, 32));
      mx = fmaxf(mx, __shfl_xor(mx, 8, 32));
      const float mnew = fmaxf(mrow[r], mx);
      cf[r] = __expf(mrow[r] - mnew);
      mrow[r] = mnew;
      S0[r] = __expf(S0[r] - mnew);
      S1[r] = __expf(S1[r] - mnew);
      float ps = S0[r] + S1[r];
      ps += __shfl_xor(ps, 1, 32);
      ps += __shfl_xor(ps, 2, 32);
      ps += __shfl_xor(ps, 4, 32);
      ps += __shfl_xor(ps, 8, 32);
      lrow[r] = lrow[r] * cf[r] + ps;
    }

    // ---- reshape P: C-layout -> A-layout via per-wave LDS tile ----
#pragma unroll
    for (int r = 0; r < 8; ++r) {
      const int row = hi * 8 + r;
      P[row * 32 + nn]      = (_Float16)S0[r];
      P[row * 32 + 16 + nn] = (_Float16)S1[r];
    }
    asm volatile("s_wait_dscnt 0" ::: "memory");
    hv16 PA;
    {
      const int m = lane & 15;
      const int c = hi * 16;
      PA.h[0] = *(const v8h*)&P[m * 32 + c];
      PA.h[1] = *(const v8h*)&P[m * 32 + 8 + c];
    }

    // ---- O = O*cf + P @ V ----
#pragma unroll
    for (int r = 0; r < 8; ++r) {
      O0[r] *= cf[r]; O1[r] *= cf[r]; O2[r] *= cf[r]; O3[r] *= cf[r];
    }
    O0 = __builtin_amdgcn_wmma_f32_16x16x32_f16(false, PA.v, false, VB0,
                                                (short)0, O0, false, false);
    O1 = __builtin_amdgcn_wmma_f32_16x16x32_f16(false, PA.v, false, VB1,
                                                (short)0, O1, false, false);
    O2 = __builtin_amdgcn_wmma_f32_16x16x32_f16(false, PA.v, false, VB2,
                                                (short)0, O2, false, false);
    O3 = __builtin_amdgcn_wmma_f32_16x16x32_f16(false, PA.v, false, VB3,
                                                (short)0, O3, false, false);
  }

  // ---- finalize: divide by row sums, store f32 output [B,F,H*DH] ----
#pragma unroll
  for (int r = 0; r < 8; ++r) {
    const float inv = 1.0f / lrow[r];
    const int row = fbase + hi * 8 + r;
    float* orow = out + ((size_t)b * F_ + row) * HD_ + h * DH_;
    orow[0 * 16 + nn] = O0[r] * inv;
    orow[1 * 16 + nn] = O1[r] * inv;
    orow[2 * 16 + nn] = O2[r] * inv;
    orow[3 * 16 + nn] = O3[r] * inv;
  }
}

// ---------------------------------------------------------------------------
extern "C" void kernel_launch(void* const* d_in, const int* in_sizes, int n_in,
                              void* d_out, int out_size, void* d_ws,
                              size_t ws_size, hipStream_t stream) {
  (void)in_sizes; (void)n_in; (void)out_size; (void)ws_size;
  const float* from_t = (const float*)d_in[0];
  const float* to_t   = (const float*)d_in[1];
  const int*   mask   = (const int*)d_in[2];
  const float* Wq = (const float*)d_in[3];
  const float* bq = (const float*)d_in[4];
  const float* Wk = (const float*)d_in[5];
  const float* bk = (const float*)d_in[6];
  const float* Wv = (const float*)d_in[7];
  const float* bv = (const float*)d_in[8];
  float* out = (float*)d_out;

  _Float16* Qh = (_Float16*)d_ws;                       // 16 MB
  _Float16* Kh = Qh + (size_t)B_ * H_ * F_ * DH_;       // 16 MB
  _Float16* Vt = Kh + (size_t)B_ * H_ * T_ * DH_;       // 16 MB

  dim3 g1(HD_ / 128, (B_ * F_) / 64, 3);
  qkv_proj_kernel<<<g1, dim3(256), 0, stream>>>(
      from_t, to_t, Wq, bq, Wk, bk, Wv, bv, Qh, Kh, Vt);

  dim3 g2(F_ / 128, B_ * H_);
  flash_attn_kernel<<<g2, dim3(256), 0, stream>>>(Qh, Kh, Vt, mask, out);
}
